// LSTMMambaBranch_87179246174648
// MI455X (gfx1250) — compile-verified, matrix operand order checked by reading, not run on hardware
//
#include <hip/hip_runtime.h>
#include <hip/hip_bf16.h>

// ---------------------------------------------------------------------------
// Types for CDNA5 WMMA (wave32, 16x16x32 bf16 -> f32)
// ---------------------------------------------------------------------------
typedef __attribute__((ext_vector_type(16))) __bf16 bf16x16;
typedef __attribute__((ext_vector_type(8)))  float  f32x8;

static_assert(sizeof(bf16x16) == 32, "bf16x16 must be 8 dwords");

#define BB   16
#define CIN  64
#define LL   512
#define HH   256
#define DM   512
#define DI   1024
#define DS   16
#define DC   4
#define DTR  32

__device__ __forceinline__ float sigmoid_f(float x) { return 1.f / (1.f + __expf(-x)); }
__device__ __forceinline__ float silu_f(float x)    { return x / (1.f + __expf(-x)); }
__device__ __forceinline__ float softplus_f(float x){ return (x > 20.f) ? x : log1pf(__expf(x)); }

// CDNA5 async global->LDS copy (16B per lane), tracked by ASYNCcnt.
__device__ __forceinline__ void async_copy16(unsigned lds_off, const void* src) {
    asm volatile("global_load_async_to_lds_b128 %0, %1, off"
                 :: "v"(lds_off), "v"(src) : "memory");
}
template <int NREM>
__device__ __forceinline__ void wait_async() {
    asm volatile("s_wait_asynccnt %0" :: "n"(NREM) : "memory");
}

// ---------------------------------------------------------------------------
// Tiled GEMM:  C[M,N] = A[M,K] @ W[N,K]^T + bias1 + bias2
// A, W are bf16. Outputs: optional f32 Cf and/or bf16 Cb.
// Block = 256 threads (8 waves). Block tile 64x128, wave tile 32x32 (2x2 WMMA),
// K chunk 32. Double-buffered async global->LDS staging (ASYNCcnt pipelined:
// issue chunk i+1, s_wait_asynccnt 3 -> chunk i landed; counts kept uniform
// per wave by clamping OOB rows instead of predicating the copies).
// ---------------------------------------------------------------------------
__global__ __launch_bounds__(256) void gemm_bf16_wmma(
    const __bf16* __restrict__ A, const __bf16* __restrict__ W,
    const float* __restrict__ bias1, const float* __restrict__ bias2,
    float* __restrict__ Cf, __bf16* __restrict__ Cb,
    int M, int N, int K)
{
    __shared__ __align__(16) __bf16 As[2][64][32];    // 2 x 4 KB
    __shared__ __align__(16) __bf16 Ws[2][128][32];   // 2 x 8 KB

    const int m0   = blockIdx.x * 64;
    const int n0   = blockIdx.y * 128;
    const int tid  = threadIdx.x;
    const int wave = tid >> 5;
    const int lane = tid & 31;
    const int kh   = lane >> 4;          // lane half selects K sub-offset
    const int ln   = lane & 15;
    const int wr   = (wave >> 2) * 32;   // wave row offset (0/32)
    const int wc   = (wave & 3) * 32;    // wave col offset (0..96)

    const int srow = tid >> 2, seg = tid & 3;    // staging: 4 16B-segments/row
    const int am  = min(m0 + srow, M - 1);
    const int wn0 = min(n0 + srow, N - 1);
    const int wn1 = min(n0 + srow + 64, N - 1);

    f32x8 acc[2][2] = {};

    auto stage = [&](int buf, int k0) {
        async_copy16((unsigned)(uintptr_t)&As[buf][srow][seg * 8],
                     &A[(size_t)am * K + k0 + seg * 8]);
        async_copy16((unsigned)(uintptr_t)&Ws[buf][srow][seg * 8],
                     &W[(size_t)wn0 * K + k0 + seg * 8]);
        async_copy16((unsigned)(uintptr_t)&Ws[buf][srow + 64][seg * 8],
                     &W[(size_t)wn1 * K + k0 + seg * 8]);
    };

    const int nchunks = K >> 5;
    stage(0, 0);
    for (int ci = 0; ci < nchunks; ++ci) {
        const int cur = ci & 1;
        if (ci + 1 < nchunks) {
            stage(cur ^ 1, (ci + 1) << 5);   // prefetch next chunk
            wait_async<3>();                 // in-order: chunk ci has landed
        } else {
            wait_async<0>();
        }
        __syncthreads();

        union { bf16x16 v; uint4 q[2]; } a[2], b[2];
        #pragma unroll
        for (int s = 0; s < 2; ++s) {
            const __bf16* ap = &As[cur][wr + s * 16 + ln][kh * 8];
            a[s].q[0] = *(const uint4*)ap;
            a[s].q[1] = *(const uint4*)(ap + 16);
            const __bf16* bp = &Ws[cur][wc + s * 16 + ln][kh * 8];
            b[s].q[0] = *(const uint4*)bp;
            b[s].q[1] = *(const uint4*)(bp + 16);
        }
        #pragma unroll
        for (int i = 0; i < 2; ++i)
            #pragma unroll
            for (int jj = 0; jj < 2; ++jj)
                acc[i][jj] = __builtin_amdgcn_wmma_f32_16x16x32_bf16(
                    false, a[i].v, false, b[jj].v, (short)0, acc[i][jj], false, false);
        __syncthreads();   // all waves done reading buf `cur` before re-staging it
    }

    #pragma unroll
    for (int i = 0; i < 2; ++i) {
        #pragma unroll
        for (int jj = 0; jj < 2; ++jj) {
            int col = n0 + wc + jj * 16 + ln;
            if (col < N) {
                float bv = (bias1 ? bias1[col] : 0.f) + (bias2 ? bias2[col] : 0.f);
                #pragma unroll
                for (int v = 0; v < 8; ++v) {
                    int row = m0 + wr + i * 16 + v + 8 * kh;
                    if (row < M) {
                        float y = acc[i][jj][v] + bv;
                        if (Cf) Cf[(size_t)row * N + col] = y;
                        if (Cb) Cb[(size_t)row * N + col] = (__bf16)y;
                    }
                }
            }
        }
    }
}

// ---------------------------------------------------------------------------
// Persistent LSTM recurrence. One workgroup, 512 threads = 16 waves.
// Wave w owns hidden block [16w,16w+16) for ALL 4 gates -> cell state c stays
// in wave registers for the whole sequence. h exchanged through 8KB bf16 LDS.
// gx values held in registers so their loads overlap the 32-WMMA chain.
// ---------------------------------------------------------------------------
__global__ __launch_bounds__(512) void lstm_scan(
    const float* __restrict__ gx,      // (L*B, 4H) token-major
    const __bf16* __restrict__ whh,    // (4H, H) bf16
    __bf16* __restrict__ hseq)         // (L*B, H) bf16
{
    __shared__ __align__(16) __bf16 hs[BB][HH];   // 8 KB

    const int tid  = threadIdx.x;
    const int wave = tid >> 5;         // hidden block
    const int lane = tid & 31;
    const int kh   = lane >> 4;
    const int ln   = lane & 15;
    const int j    = wave * 16 + ln;

    for (int i = tid; i < BB * HH; i += 512) ((__bf16*)hs)[i] = (__bf16)0.f;
    f32x8 c = {};
    __syncthreads();

    for (int l = 0; l < LL; ++l) {
        const float* grow = gx + (size_t)l * BB * (4 * HH);

        // issue gx loads now; consume only after the WMMA chain
        float gxv[4][8];
        #pragma unroll
        for (int g = 0; g < 4; ++g) {
            int n = g * HH + j;
            #pragma unroll
            for (int v = 0; v < 8; ++v)
                gxv[g][v] = grow[(v + 8 * kh) * (4 * HH) + n];
        }
        // prefetch next step's gx row into caches
        if (l + 1 < LL)
            __builtin_prefetch(grow + (size_t)BB * 4 * HH + tid * 32, 0, 0);

        f32x8 acc[4] = {};
        #pragma unroll 1
        for (int kc = 0; kc < HH / 32; ++kc) {
            union { bf16x16 v; uint4 q[2]; } fa;
            const __bf16* hp = &hs[ln][kc * 32 + kh * 8];
            fa.q[0] = *(const uint4*)hp;
            fa.q[1] = *(const uint4*)(hp + 16);

            #pragma unroll
            for (int g = 0; g < 4; ++g) {
                union { bf16x16 v; uint4 q[2]; } fb;
                const __bf16* wrow = whh + (size_t)(g * HH + j) * HH + kc * 32 + kh * 8;
                fb.q[0] = *(const uint4*)wrow;
                fb.q[1] = *(const uint4*)(wrow + 16);
                acc[g] = __builtin_amdgcn_wmma_f32_16x16x32_bf16(
                    false, fa.v, false, fb.v, (short)0, acc[g], false, false);
            }
        }
        __syncthreads();   // all waves done reading hs

        __bf16* hrow = hseq + (size_t)l * BB * HH;
        #pragma unroll
        for (int v = 0; v < 8; ++v) {
            int b  = v + 8 * kh;
            float ig = sigmoid_f(acc[0][v] + gxv[0][v]);
            float fg = sigmoid_f(acc[1][v] + gxv[1][v]);
            float gg = tanhf(acc[2][v] + gxv[2][v]);
            float og = sigmoid_f(acc[3][v] + gxv[3][v]);
            float cv = fg * c[v] + ig * gg;
            c[v] = cv;
            float hv = og * tanhf(cv);
            __bf16 hb = (__bf16)hv;
            hs[b][j] = hb;
            hrow[b * HH + j] = hb;
        }
        __syncthreads();
    }
}

// ---------------------------------------------------------------------------
// Small helper kernels
// ---------------------------------------------------------------------------
__global__ void f32_to_bf16_k(const float* __restrict__ in, __bf16* __restrict__ out, int n) {
    int i = blockIdx.x * 256 + threadIdx.x;
    if (i < n) out[i] = (__bf16)in[i];
}

// x (B,CIN,L) -> xs token-major (L*B, CIN), bf16
__global__ void transpose_x_k(const float* __restrict__ x, __bf16* __restrict__ xs) {
    int i = blockIdx.x * 256 + threadIdx.x;
    if (i >= LL * BB * CIN) return;
    int ci = i % CIN; int t = i / CIN; int b = t % BB; int l = t / BB;
    xs[i] = (__bf16)x[((size_t)b * CIN + ci) * LL + l];
}

// causal depthwise conv (DC=4) + SiLU; writes f32 (for scan) and bf16 (for GEMM)
__global__ void conv_silu_k(const float* __restrict__ xz, const float* __restrict__ cw,
                            const float* __restrict__ cb,
                            float* __restrict__ uf, __bf16* __restrict__ ub)
{
    int i = blockIdx.x * 256 + threadIdx.x;       // over L*B*DI
    if (i >= LL * BB * DI) return;
    int d = i % DI; int t = i / DI; int b = t % BB; int l = t / BB;
    float acc = cb[d];
    #pragma unroll
    for (int jj = 0; jj < DC; ++jj) {
        int ls = l - (DC - 1) + jj;
        if (ls >= 0) acc += cw[d * DC + jj] * xz[((size_t)(ls * BB + b)) * (2 * DI) + d];
    }
    float val = silu_f(acc);
    uf[i] = val;
    ub[i] = (__bf16)val;
}

// Selective scan. Block covers 256 consecutive d for ONE b (b uniform per block
// -> dbl/dt row loads are scalar). dt_proj + softplus fused. 16 states in regs.
__global__ __launch_bounds__(256) void sel_scan_k(
    const float* __restrict__ dbl,     // (L*B, 64): dt[0:32], B[32:48], C[48:64]
    const float* __restrict__ u,       // (L*B, DI) f32
    const float* __restrict__ xz,      // (L*B, 2*DI), res at [DI:2DI]
    const float* __restrict__ dt_w,    // (DI, DTR)
    const float* __restrict__ dt_b,
    const float* __restrict__ A_log,   // (DI, DS)
    const float* __restrict__ Dp,
    __bf16* __restrict__ ys)           // (L*B, DI) bf16
{
    const int b = blockIdx.x >> 2;                       // 4 blocks per batch
    const int d = ((blockIdx.x & 3) << 8) | threadIdx.x; // 0..1023

    float Arow[DS], h[DS], wdt[DTR];
    #pragma unroll
    for (int n = 0; n < DS; ++n) { Arow[n] = -__expf(A_log[d * DS + n]); h[n] = 0.f; }
    #pragma unroll
    for (int r = 0; r < DTR; ++r) wdt[r] = dt_w[d * DTR + r];
    const float bd = dt_b[d], Dd = Dp[d];

    for (int l = 0; l < LL; ++l) {
        int t = l * BB + b;
        const float* row = dbl + (size_t)t * 64;
        float dtv = bd;
        #pragma unroll
        for (int r = 0; r < DTR; ++r) dtv += row[r] * wdt[r];
        float delta = softplus_f(dtv);
        float ut = u[(size_t)t * DI + d];
        if (l + 1 < LL) __builtin_prefetch(&u[(size_t)(t + BB) * DI + d], 0, 0);
        float du = delta * ut;
        float y = 0.f;
        #pragma unroll
        for (int n = 0; n < DS; ++n) {
            h[n] = __expf(delta * Arow[n]) * h[n] + du * row[32 + n];
            y += h[n] * row[48 + n];
        }
        float resv = xz[(size_t)t * (2 * DI) + DI + d];
        ys[(size_t)t * DI + d] = (__bf16)((y + ut * Dd) * silu_f(resv));
    }
}

// Per-token LayerNorm + mean over L. One block per batch b, 256 threads.
__global__ __launch_bounds__(256) void norm_pool_k(
    const float* __restrict__ seq,     // (L*B, DM)
    const float* __restrict__ gam, const float* __restrict__ bet,
    float* __restrict__ out)           // (B, DM)
{
    __shared__ float r1[256], r2[256];
    __shared__ float mu_s, rs_s;
    const int b = blockIdx.x, tid = threadIdx.x;
    float a0 = 0.f, a1 = 0.f;

    for (int l = 0; l < LL; ++l) {
        const float* row = seq + ((size_t)l * BB + b) * DM;
        float v0 = row[tid], v1 = row[tid + 256];
        r1[tid] = v0 + v1;
        r2[tid] = v0 * v0 + v1 * v1;
        __syncthreads();
        for (int off = 128; off > 0; off >>= 1) {
            if (tid < off) { r1[tid] += r1[tid + off]; r2[tid] += r2[tid + off]; }
            __syncthreads();
        }
        if (tid == 0) {
            float mu = r1[0] / (float)DM;
            mu_s = mu;
            rs_s = rsqrtf(r2[0] / (float)DM - mu * mu + 1e-5f);
        }
        __syncthreads();
        a0 += (v0 - mu_s) * rs_s;
        a1 += (v1 - mu_s) * rs_s;
        __syncthreads();
    }
    out[(size_t)b * DM + tid]       = (a0 / (float)LL) * gam[tid]       + bet[tid];
    out[(size_t)b * DM + tid + 256] = (a1 / (float)LL) * gam[tid + 256] + bet[tid + 256];
}

// ---------------------------------------------------------------------------
// Host launcher
// ---------------------------------------------------------------------------
extern "C" void kernel_launch(void* const* d_in, const int* in_sizes, int n_in,
                              void* d_out, int out_size, void* d_ws, size_t ws_size,
                              hipStream_t stream)
{
    const float* x        = (const float*)d_in[0];
    const float* w_ih0    = (const float*)d_in[1];
    const float* w_hh0    = (const float*)d_in[2];
    const float* b_ih0    = (const float*)d_in[3];
    const float* b_hh0    = (const float*)d_in[4];
    const float* w_ih1    = (const float*)d_in[5];
    const float* w_hh1    = (const float*)d_in[6];
    const float* b_ih1    = (const float*)d_in[7];
    const float* b_hh1    = (const float*)d_in[8];
    const float* proj_w   = (const float*)d_in[9];
    const float* proj_b   = (const float*)d_in[10];
    const float* in_proj_w= (const float*)d_in[11];
    const float* conv_w   = (const float*)d_in[12];
    const float* conv_b   = (const float*)d_in[13];
    const float* x_proj_w = (const float*)d_in[14];
    const float* dt_proj_w= (const float*)d_in[15];
    const float* dt_proj_b= (const float*)d_in[16];
    const float* A_log    = (const float*)d_in[17];
    const float* Dp       = (const float*)d_in[18];
    const float* out_proj_w=(const float*)d_in[19];
    const float* norm_g   = (const float*)d_in[20];
    const float* norm_b   = (const float*)d_in[21];

    const int T = LL * BB;            // 8192 tokens
    char* ws = (char*)d_ws;
    size_t o = 0;
    auto alloc_f = [&](size_t n) { float*  p = (float*) (ws + o); o += n * 4; return p; };
    auto alloc_b = [&](size_t n) { __bf16* p = (__bf16*)(ws + o); o += ((n * 2 + 15) & ~(size_t)15); return p; };

    __bf16* xs_b   = alloc_b((size_t)T * CIN);
    float*  gx     = alloc_f((size_t)T * 4 * HH);
    __bf16* h1_b   = alloc_b((size_t)T * HH);
    __bf16* h2_b   = alloc_b((size_t)T * HH);
    __bf16* seq_b  = alloc_b((size_t)T * DM);
    float*  xz     = alloc_f((size_t)T * 2 * DI);
    float*  u_f    = alloc_f((size_t)T * DI);
    __bf16* u_b    = alloc_b((size_t)T * DI);
    float*  dbl    = alloc_f((size_t)T * 64);
    __bf16* ys_b   = alloc_b((size_t)T * DI);
    float*  seq2   = alloc_f((size_t)T * DM);
    __bf16* whh0_b = alloc_b((size_t)4 * HH * HH);
    __bf16* whh1_b = alloc_b((size_t)4 * HH * HH);
    __bf16* wih0_b = alloc_b((size_t)4 * HH * CIN);
    __bf16* wih1_b = alloc_b((size_t)4 * HH * HH);
    __bf16* wproj_b= alloc_b((size_t)DM * HH);
    __bf16* winp_b = alloc_b((size_t)2 * DI * DM);
    __bf16* wxp_b  = alloc_b((size_t)(DTR + 2 * DS) * DI);
    __bf16* wout_b = alloc_b((size_t)DM * DI);
    (void)ws_size; (void)n_in; (void)in_sizes; (void)out_size;

    auto cvt = [&](const float* src, __bf16* dst, int n) {
        f32_to_bf16_k<<<(n + 255) / 256, 256, 0, stream>>>(src, dst, n);
    };
    cvt(w_hh0, whh0_b, 4*HH*HH);
    cvt(w_hh1, whh1_b, 4*HH*HH);
    cvt(w_ih0, wih0_b, 4*HH*CIN);
    cvt(w_ih1, wih1_b, 4*HH*HH);
    cvt(proj_w, wproj_b, DM*HH);
    cvt(in_proj_w, winp_b, 2*DI*DM);
    cvt(x_proj_w, wxp_b, (DTR + 2*DS)*DI);
    cvt(out_proj_w, wout_b, DM*DI);

    transpose_x_k<<<(T*CIN + 255)/256, 256, 0, stream>>>(x, xs_b);

    auto gemm = [&](const __bf16* A, const __bf16* W, const float* bb1, const float* bb2,
                    float* Cf, __bf16* Cb, int M, int N, int K) {
        dim3 grid((M + 63) / 64, (N + 127) / 128);
        gemm_bf16_wmma<<<grid, 256, 0, stream>>>(A, W, bb1, bb2, Cf, Cb, M, N, K);
    };

    // ---- LSTM layer 0 ----
    gemm(xs_b, wih0_b, b_ih0, b_hh0, gx, nullptr, T, 4*HH, CIN);
    lstm_scan<<<1, 512, 0, stream>>>(gx, whh0_b, h1_b);
    // ---- LSTM layer 1 ----
    gemm(h1_b, wih1_b, b_ih1, b_hh1, gx, nullptr, T, 4*HH, HH);
    lstm_scan<<<1, 512, 0, stream>>>(gx, whh1_b, h2_b);

    // ---- projection to d_model (bf16, consumed by in_proj GEMM) ----
    gemm(h2_b, wproj_b, proj_b, nullptr, nullptr, seq_b, T, DM, HH);

    // ---- Mamba block ----
    gemm(seq_b, winp_b, nullptr, nullptr, xz, nullptr, T, 2*DI, DM);
    conv_silu_k<<<(T*DI + 255)/256, 256, 0, stream>>>(xz, conv_w, conv_b, u_f, u_b);
    gemm(u_b, wxp_b, nullptr, nullptr, dbl, nullptr, T, DTR + 2*DS, DI);
    sel_scan_k<<<4 * BB, 256, 0, stream>>>(dbl, u_f, xz, dt_proj_w, dt_proj_b, A_log, Dp, ys_b);
    gemm(ys_b, wout_b, nullptr, nullptr, seq2, nullptr, T, DM, DI);

    // ---- LayerNorm + mean over L ----
    norm_pool_k<<<BB, 256, 0, stream>>>(seq2, norm_g, norm_b, (float*)d_out);
}
